// RNN_2465311228514
// MI455X (gfx1250) — matrix operand order repaired and checked by reading, not compiled, not run on hardware
//
#include <hip/hip_runtime.h>
#include <hip/hip_bf16.h>

typedef __attribute__((ext_vector_type(16))) _Float16 v16h;
typedef __attribute__((ext_vector_type(8)))  float    v8f;

#define T_STEPS 28
#define HID     128
#define IN_SZ   28
#define NCLS    10
#define BT      64     // batch rows per workgroup
#define NTHREADS 256   // 8 waves

// LDS strides padded to rotate banks: 136 halves = 272 B = 68 banks -> 4-bank
// shift per row/col, so 16 fragment lanes cover all 64 banks conflict-free.
#define LDH 136        // stride for HID-wide (128) arrays
#define LDX 40         // stride for 32-wide (padded 28) arrays

#if __has_builtin(__builtin_amdgcn_tanhf)
#define TANH(v) __builtin_amdgcn_tanhf(v)    // V_TANH_F32 (CDNA5 TRANS op)
#else
#define TANH(v) tanhf(v)
#endif

struct SharedStore {
    _Float16 Wih0[HID * LDX];    // ~10 KB (K padded 28->32, then to LDX)
    _Float16 Whh0[HID * LDH];    // ~34 KB
    _Float16 Wih1[HID * LDH];    // ~34 KB
    _Float16 Whh1[HID * LDH];    // ~34 KB
    _Float16 fcw [16  * LDH];    // ~4.25 KB (rows padded 10->16)
    _Float16 h0  [BT  * LDH];    // ~17 KB
    _Float16 h1  [BT  * LDH];    // ~17 KB
    _Float16 xt  [BT  * LDX];    // ~5 KB
    float    b0[HID];
    float    b1[HID];
    float    fcb[16];
};                               // ~156 KB total

// ---- fragment loaders per CDNA5 ISA 7.12.2 (wave32) ----
// A: 16x32 f16 tile (MxK). lane<16: row=M=lane, K chunks {kb..kb+7, kb+16..kb+23}, kb=k0
//    lane>=16: same row, kb=k0+8.
__device__ __forceinline__ v16h load_frag_A(const _Float16* base, int ld,
                                            int m0, int k0, int lane) {
    int row = m0 + (lane & 15);
    int kb  = k0 + ((lane >> 4) << 3);          // +0 or +8
    const _Float16* p = base + row * ld + kb;
    union { v16h v; uint4 q[2]; } u;
    u.q[0] = *(const uint4*)(p);                // K = kb .. kb+7
    u.q[1] = *(const uint4*)(p + 16);           // K = kb+16 .. kb+23
    return u.v;
}

// B: 32x16 f16 tile (KxN) with B[k][n] = W[n0+n][k] (computing h @ W^T).
// lane<16: col N=lane, K chunks {kb..kb+7, kb+16..kb+23}, kb=k0; lane>=16: col N=lane-16, kb=k0+8.
__device__ __forceinline__ v16h load_frag_B(const _Float16* Wrow, int ld,
                                            int k0, int n0, int lane) {
    int col = n0 + (lane & 15);
    int kb  = k0 + ((lane >> 4) << 3);          // +0 or +8
    const _Float16* p = Wrow + col * ld + kb;
    union { v16h v; uint4 q[2]; } u;
    u.q[0] = *(const uint4*)(p);
    u.q[1] = *(const uint4*)(p + 16);
    return u.v;
}

// D/C layout: lane gives N = lane&15; VGPR r gives M = r + (lane<16 ? 0 : 8).
__device__ __forceinline__ void store_tile_tanh(_Float16* h, int ld, int m0, int n0,
                                                int lane, const v8f acc, const float* bias) {
    int n  = n0 + (lane & 15);
    int mo = m0 + ((lane >> 4) << 3);
    float bv = bias[n];
#pragma unroll
    for (int r = 0; r < 8; ++r)
        h[(mo + r) * ld + n] = (_Float16)TANH(acc[r] + bv);
}

__device__ __forceinline__ void cvt_w(_Float16* dst, const float* src,
                                      int rows, int cols, int ldd, int tid) {
    for (int i = tid; i < rows * ldd; i += NTHREADS) {
        int r = i / ldd, c = i - r * ldd;
        dst[i] = (c < cols) ? (_Float16)src[r * cols + c] : (_Float16)0.0f;
    }
}

#define WMMA_F16(a, b, c) \
    __builtin_amdgcn_wmma_f32_16x16x32_f16(false, (a), false, (b), (short)0, (c), false, false)

__global__ __launch_bounds__(NTHREADS) void rnn2_fused_kernel(
    const float* __restrict__ x,
    const float* __restrict__ Wih0, const float* __restrict__ Whh0,
    const float* __restrict__ bih0, const float* __restrict__ bhh0,
    const float* __restrict__ Wih1, const float* __restrict__ Whh1,
    const float* __restrict__ bih1, const float* __restrict__ bhh1,
    const float* __restrict__ fcw,  const float* __restrict__ fcb,
    float* __restrict__ out) {

    __shared__ SharedStore S;
    const int tid   = threadIdx.x;
    const int lane  = tid & 31;
    const int wv    = tid >> 5;           // 0..7
    const int n0    = wv * 16;            // this wave's N column block
    const int bBase = blockIdx.x * BT;

    // ---- stage weights into LDS as f16 (zero-padded), fold biases ----
    cvt_w(S.Wih0, Wih0, HID, IN_SZ, LDX, tid);
    cvt_w(S.Whh0, Whh0, HID, HID,   LDH, tid);
    cvt_w(S.Wih1, Wih1, HID, HID,   LDH, tid);
    cvt_w(S.Whh1, Whh1, HID, HID,   LDH, tid);
    for (int i = tid; i < 16 * LDH; i += NTHREADS) {
        int r = i / LDH, c = i - r * LDH;
        S.fcw[i] = (r < NCLS && c < HID) ? (_Float16)fcw[r * HID + c] : (_Float16)0.0f;
    }
    for (int i = tid; i < HID; i += NTHREADS) {
        S.b0[i] = bih0[i] + bhh0[i];
        S.b1[i] = bih1[i] + bhh1[i];
    }
    if (tid < 16) S.fcb[tid] = (tid < NCLS) ? fcb[tid] : 0.0f;
    for (int i = tid; i < BT * LDH; i += NTHREADS) {
        S.h0[i] = (_Float16)0.0f;
        S.h1[i] = (_Float16)0.0f;
    }
    __syncthreads();

    // ---- hoist loop-invariant B fragments (weights) into VGPRs ----
    // 13 fragments * 8 VGPRs = 104 VGPRs, resident across the whole t-loop.
    v16h bWih0 = load_frag_B(S.Wih0, LDX, 0, n0, lane);
    v16h bWhh0[4], bWih1[4], bWhh1[4];
#pragma unroll
    for (int ks = 0; ks < 4; ++ks) {
        bWhh0[ks] = load_frag_B(S.Whh0, LDH, ks * 32, n0, lane);
        bWih1[ks] = load_frag_B(S.Wih1, LDH, ks * 32, n0, lane);
        bWhh1[ks] = load_frag_B(S.Whh1, LDH, ks * 32, n0, lane);
    }

    const v8f vzero = {0.f, 0.f, 0.f, 0.f, 0.f, 0.f, 0.f, 0.f};

    for (int t = 0; t < T_STEPS; ++t) {
        // ---- stream x_t tile HBM -> LDS (f32 -> f16, pad K) ----
        for (int i = tid; i < BT * 32; i += NTHREADS) {
            int r = i >> 5, c = i & 31;
            float v = (c < IN_SZ)
                ? x[((size_t)(bBase + r) * T_STEPS + t) * IN_SZ + c] : 0.0f;
            S.xt[r * LDX + c] = (_Float16)v;
        }
        // prefetch next step's x tile while this step computes
        if (t + 1 < T_STEPS && tid < BT)
            __builtin_prefetch(&x[((size_t)(bBase + tid) * T_STEPS + (t + 1)) * IN_SZ], 0, 1);
        __syncthreads();

        // ---- layer 0: acc = x_t @ Wih0^T + h0 @ Whh0^T ----
        v8f acc[4];
#pragma unroll
        for (int mt = 0; mt < 4; ++mt) {
            v16h aX = load_frag_A(S.xt, LDX, mt * 16, 0, lane);
            acc[mt] = WMMA_F16(aX, bWih0, vzero);
            v16h a0 = load_frag_A(S.h0, LDH, mt * 16, 0,  lane);
            v16h a1 = load_frag_A(S.h0, LDH, mt * 16, 32, lane);
            v16h a2 = load_frag_A(S.h0, LDH, mt * 16, 64, lane);
            v16h a3 = load_frag_A(S.h0, LDH, mt * 16, 96, lane);
            acc[mt] = WMMA_F16(a0, bWhh0[0], acc[mt]);
            acc[mt] = WMMA_F16(a1, bWhh0[1], acc[mt]);
            acc[mt] = WMMA_F16(a2, bWhh0[2], acc[mt]);
            acc[mt] = WMMA_F16(a3, bWhh0[3], acc[mt]);
        }
        __syncthreads();   // all reads of h0(t-1) complete before overwrite
#pragma unroll
        for (int mt = 0; mt < 4; ++mt)
            store_tile_tanh(S.h0, LDH, mt * 16, n0, lane, acc[mt], S.b0);
        __syncthreads();   // h0(t) visible to all waves

        // ---- layer 1: acc = h0 @ Wih1^T + h1 @ Whh1^T ----
#pragma unroll
        for (int mt = 0; mt < 4; ++mt) {
            v16h a0 = load_frag_A(S.h0, LDH, mt * 16, 0,  lane);
            v16h a1 = load_frag_A(S.h0, LDH, mt * 16, 32, lane);
            v16h a2 = load_frag_A(S.h0, LDH, mt * 16, 64, lane);
            v16h a3 = load_frag_A(S.h0, LDH, mt * 16, 96, lane);
            acc[mt] = WMMA_F16(a0, bWih1[0], vzero);
            acc[mt] = WMMA_F16(a1, bWih1[1], acc[mt]);
            acc[mt] = WMMA_F16(a2, bWih1[2], acc[mt]);
            acc[mt] = WMMA_F16(a3, bWih1[3], acc[mt]);
            v16h r0 = load_frag_A(S.h1, LDH, mt * 16, 0,  lane);
            v16h r1 = load_frag_A(S.h1, LDH, mt * 16, 32, lane);
            v16h r2 = load_frag_A(S.h1, LDH, mt * 16, 64, lane);
            v16h r3 = load_frag_A(S.h1, LDH, mt * 16, 96, lane);
            acc[mt] = WMMA_F16(r0, bWhh1[0], acc[mt]);
            acc[mt] = WMMA_F16(r1, bWhh1[1], acc[mt]);
            acc[mt] = WMMA_F16(r2, bWhh1[2], acc[mt]);
            acc[mt] = WMMA_F16(r3, bWhh1[3], acc[mt]);
        }
        __syncthreads();   // all reads of h1(t-1) complete
#pragma unroll
        for (int mt = 0; mt < 4; ++mt)
            store_tile_tanh(S.h1, LDH, mt * 16, n0, lane, acc[mt], S.b1);
        __syncthreads();
    }

    // ---- FC head: out = h1_last @ fcw^T + fcb (waves 0..3, one M-tile each) ----
    if (wv < 4) {
        v8f acc = vzero;
#pragma unroll
        for (int ks = 0; ks < 4; ++ks) {
            v16h b = load_frag_B(S.fcw, LDH, ks * 32, 0, lane);
            v16h a = load_frag_A(S.h1, LDH, wv * 16, ks * 32, lane);
            acc = WMMA_F16(a, b, acc);
        }
        int n  = lane & 15;
        int mo = wv * 16 + ((lane >> 4) << 3);
        if (n < NCLS) {
            float bv = S.fcb[n];
#pragma unroll
            for (int r = 0; r < 8; ++r)
                out[(size_t)(bBase + mo + r) * NCLS + n] = acc[r] + bv;
        }
    }
}

extern "C" void kernel_launch(void* const* d_in, const int* in_sizes, int n_in,
                              void* d_out, int out_size, void* d_ws, size_t ws_size,
                              hipStream_t stream) {
    const float* x    = (const float*)d_in[0];
    const float* Wih0 = (const float*)d_in[1];
    const float* Whh0 = (const float*)d_in[2];
    const float* bih0 = (const float*)d_in[3];
    const float* bhh0 = (const float*)d_in[4];
    const float* Wih1 = (const float*)d_in[5];
    const float* Whh1 = (const float*)d_in[6];
    const float* bih1 = (const float*)d_in[7];
    const float* bhh1 = (const float*)d_in[8];
    const float* fcw  = (const float*)d_in[9];
    const float* fcb  = (const float*)d_in[10];
    float* out = (float*)d_out;

    int B = in_sizes[0] / (T_STEPS * IN_SZ);   // 8192
    int blocks = B / BT;                        // 128

    rnn2_fused_kernel<<<blocks, NTHREADS, 0, stream>>>(
        x, Wih0, Whh0, bih0, bhh0, Wih1, Whh1, bih1, bhh1, fcw, fcb, out);
}
